// SidechainDenoisingLayer_5832565588251
// MI455X (gfx1250) — compile-verified, matrix operand order checked by reading, not run on hardware
//
#include <hip/hip_runtime.h>
#include <hip/hip_bf16.h>

// ---------------------------------------------------------------------------
// CDNA5 / gfx1250 implementation of the IPMP sidechain denoising layer.
// Edge MLPs run on v_wmma_f32_16x16x32_bf16 (fp32 accumulate), 32 edges/block
// (two 16-row M-tiles) so every B-weight fragment fetched from L2 feeds two
// WMMAs.
// ---------------------------------------------------------------------------

typedef __bf16 bf16;
typedef __attribute__((ext_vector_type(16))) __bf16 v16bf;
typedef __attribute__((ext_vector_type(8)))  __bf16 v8bf;
typedef __attribute__((ext_vector_type(8)))  float  v8f;

#define NNODES   10000
#define NEDGES   160000
#define PNT      8
#define CS       128
#define CLAT     64
#define CZ       64
#define CH       128
#define HDIM     256          // CS + 2*CLAT
#define MIN_DIM  608          // 2H + CZ + 3P + P
#define EIN_DIM  576          // 2H + CZ
#define MTILE    32           // edges per block (2 WMMA M-tiles)

union V16U { v16bf v; v8bf h[2]; };

__device__ __forceinline__ v8f wmma_bf16(v16bf a, v16bf b, v8f c) {
  return __builtin_amdgcn_wmma_f32_16x16x32_bf16(false, a, false, b, (short)0, c,
                                                 false, false);
}

// A fragment from an LDS-resident bf16 row-major tile.
// rowBase points at &tile[row * rowStride]; koff = kt*32 + 8*(lane>>4).
__device__ __forceinline__ v16bf load_afrag(const bf16* rowBase, int koff) {
  V16U u;
  u.h[0] = *(const v8bf*)(rowBase + koff);
  u.h[1] = *(const v8bf*)(rowBase + koff + 16);
  return u.v;
}

// B fragment from the fragment-packed global weight buffer.
__device__ __forceinline__ v16bf load_bfrag(const bf16* frags, int fidx, int lane) {
  return *(const v16bf*)(frags + (size_t)fidx * 512 + (size_t)lane * 16);
}

// ---------------------------------------------------------------------------
// Weight repack: fp32 [K,N] row-major  ->  bf16 B-fragments.
// grid = (Ntiles, Ktiles), block = 32.  frag f = kt*Ntiles + nt holds 512 bf16:
// lane L stores col N = nt*16 + (L&15); k-locals: i<8 -> 8*(L>>4)+i,
// i>=8 -> 16 + 8*(L>>4) + (i-8)  (mirrors the 16-bit A layout).
// ---------------------------------------------------------------------------
__global__ void wprep_kernel(const float* __restrict__ w, bf16* __restrict__ out,
                             int Ncols) {
  const int nt = blockIdx.x, kt = blockIdx.y, lane = threadIdx.x;
  const int f  = kt * gridDim.x + nt;
  bf16* dst = out + (size_t)f * 512 + (size_t)lane * 16;
  const int n  = nt * 16 + (lane & 15);
  const int kb = kt * 32 + 8 * (lane >> 4);
  #pragma unroll
  for (int i = 0; i < 8; ++i) dst[i]     = (bf16)w[(size_t)(kb + i) * Ncols + n];
  #pragma unroll
  for (int i = 0; i < 8; ++i) dst[8 + i] = (bf16)w[(size_t)(kb + 16 + i) * Ncols + n];
}

__global__ void zero_kernel(float* __restrict__ p, int n) {
  int i = blockIdx.x * blockDim.x + threadIdx.x;
  if (i < n) p[i] = 0.0f;
}

// ---------------------------------------------------------------------------
// Per-node prep: s = [node | latent | 0], local points (s @ points_w + b),
// global-frame points.  One wave per node.
// ---------------------------------------------------------------------------
__global__ void node_prep_kernel(const float* __restrict__ latf,
                                 const float* __restrict__ nodef,
                                 const float* __restrict__ rots,
                                 const float* __restrict__ trans,
                                 const float* __restrict__ pw,
                                 const float* __restrict__ pb,
                                 float* __restrict__ s,
                                 float* __restrict__ pts,
                                 float* __restrict__ glob) {
  const int n = blockIdx.x;
  const int lane = threadIdx.x;
  __shared__ float sp[PNT * 3];

  for (int i = lane; i < CS; i += 32)
    s[(size_t)n * HDIM + i] = nodef[(size_t)n * CS + i];
  for (int i = lane; i < CLAT; i += 32) {
    s[(size_t)n * HDIM + CS + i]        = latf[(size_t)n * CLAT + i];
    s[(size_t)n * HDIM + CS + CLAT + i] = 0.0f;
  }
  if (lane < PNT * 3) {
    float a = pb[lane];
    for (int k = 0; k < CS; ++k)
      a += nodef[(size_t)n * CS + k] * pw[k * (PNT * 3) + lane];
    for (int k = 0; k < CLAT; ++k)
      a += latf[(size_t)n * CLAT + k] * pw[(CS + k) * (PNT * 3) + lane];
    pts[(size_t)n * PNT * 3 + lane] = a;
    sp[lane] = a;
  }
  __syncthreads();
  if (lane < PNT) {
    const float* R = rots + (size_t)n * 9;
    const float* t = trans + (size_t)n * 3;
    const float px = sp[lane * 3 + 0], py = sp[lane * 3 + 1], pz = sp[lane * 3 + 2];
    float* g = glob + (size_t)n * PNT * 3 + lane * 3;
    g[0] = R[0] * px + R[1] * py + R[2] * pz + t[0];
    g[1] = R[3] * px + R[4] * py + R[5] * pz + t[1];
    g[2] = R[6] * px + R[7] * py + R[8] * pz + t[2];
  }
}

// ---------------------------------------------------------------------------
// Message MLP over edges. 1 block = 32 edges, 128 threads (4 waves).
// Stages m_in[32][608] as bf16 in LDS, then three WMMA GEMM layers
// (two M-tiles per wave so each B fragment feeds 2 WMMAs), then scatters
// msg*ew into agg via fp32 atomics.
// ---------------------------------------------------------------------------
__launch_bounds__(128)
__global__ void msg_kernel(const float* __restrict__ s,
                           const float* __restrict__ pts,
                           const float* __restrict__ glob,
                           const float* __restrict__ ef,
                           const int*   __restrict__ ei,
                           const float* __restrict__ rots,
                           const float* __restrict__ trans,
                           const float* __restrict__ mask,
                           const bf16*  __restrict__ w1f, const float* __restrict__ b1,
                           const bf16*  __restrict__ w2f, const float* __restrict__ b2,
                           const bf16*  __restrict__ w3f, const float* __restrict__ b3,
                           float* __restrict__ agg, float* __restrict__ deg) {
  __shared__ alignas(16) bf16 smA[MTILE * MIN_DIM];
  __shared__ alignas(16) bf16 smH1[MTILE * CH];
  __shared__ alignas(16) bf16 smH2[MTILE * CH];
  __shared__ int   smDst[MTILE];
  __shared__ float smEw[MTILE];

  const int tid  = threadIdx.x;
  const int lane = tid & 31;
  const int wave = tid >> 5;
  const int e0   = blockIdx.x * MTILE;

  // ---- stage m_in = [s[dst] | s[src] | z | q | dist] as bf16 (4 thr/edge) ----
  {
    const int le = tid >> 2, sub = tid & 3;
    const int e = e0 + le;
    const int src = ei[e], dst = ei[NEDGES + e];
    bf16* row = smA + (size_t)le * MIN_DIM;
    const float* sd = s + (size_t)dst * HDIM;
    const float* ss = s + (size_t)src * HDIM;
    for (int i = sub * 64; i < sub * 64 + 64; ++i) {
      row[i]        = (bf16)sd[i];
      row[HDIM + i] = (bf16)ss[i];
    }
    for (int i = sub * 16; i < sub * 16 + 16; ++i)
      row[2 * HDIM + i] = (bf16)ef[(size_t)e * CZ + i];
    // points p = 2*sub, 2*sub+1: q = R[dst]^T (glob[src,p]-t[dst]); dist
    const float* Rd = rots  + (size_t)dst * 9;
    const float* td = trans + (size_t)dst * 3;
    #pragma unroll
    for (int pi = 0; pi < 2; ++pi) {
      const int p = sub * 2 + pi;
      const float* gp = glob + ((size_t)src * PNT + p) * 3;
      const float rx = gp[0] - td[0], ry = gp[1] - td[1], rz = gp[2] - td[2];
      const float q0 = Rd[0] * rx + Rd[3] * ry + Rd[6] * rz;
      const float q1 = Rd[1] * rx + Rd[4] * ry + Rd[7] * rz;
      const float q2 = Rd[2] * rx + Rd[5] * ry + Rd[8] * rz;
      const float* pp = pts + ((size_t)dst * PNT + p) * 3;
      const float d0 = q0 - pp[0], d1 = q1 - pp[1], d2 = q2 - pp[2];
      const float dist = sqrtf(d0 * d0 + d1 * d1 + d2 * d2 + 1e-8f);
      row[2 * HDIM + CZ + p * 3 + 0] = (bf16)q0;
      row[2 * HDIM + CZ + p * 3 + 1] = (bf16)q1;
      row[2 * HDIM + CZ + p * 3 + 2] = (bf16)q2;
      row[2 * HDIM + CZ + PNT * 3 + p] = (bf16)dist;
    }
  }
  if (tid < MTILE) {
    const int e = e0 + tid;
    const int src = ei[e], dst = ei[NEDGES + e];
    const float ew = mask[src] * mask[dst];
    smDst[tid] = dst;
    smEw[tid]  = ew;
    atomicAdd(deg + dst, ew);
  }
  __syncthreads();

  const int nlo = lane & 15;
  const int hi8 = (lane >> 4) * 8;
  // per-M-tile A row bases
  const bf16* rA1[2] = { smA  + (size_t)nlo * MIN_DIM,
                         smA  + (size_t)(nlo + 16) * MIN_DIM };
  const bf16* rA2[2] = { smH1 + (size_t)nlo * CH,
                         smH1 + (size_t)(nlo + 16) * CH };
  const bf16* rA3[2] = { smH2 + (size_t)nlo * CH,
                         smH2 + (size_t)(nlo + 16) * CH };

  // ---- layer 1: [32,608] @ [608,128]; wave: 2 n-tiles x 2 m-tiles ----
  {
    v8f acc[2][2];
    #pragma unroll
    for (int j = 0; j < 2; ++j) {
      const float bv = b1[(wave * 2 + j) * 16 + nlo];
      #pragma unroll
      for (int m = 0; m < 2; ++m)
        #pragma unroll
        for (int r = 0; r < 8; ++r) acc[m][j][r] = bv;
    }
    for (int kt = 0; kt < MIN_DIM / 32; ++kt) {
      const int koff = kt * 32 + hi8;
      const v16bf a0 = load_afrag(rA1[0], koff);
      const v16bf a1 = load_afrag(rA1[1], koff);
      #pragma unroll
      for (int j = 0; j < 2; ++j) {
        const v16bf b = load_bfrag(w1f, kt * 8 + wave * 2 + j, lane);
        acc[0][j] = wmma_bf16(a0, b, acc[0][j]);
        acc[1][j] = wmma_bf16(a1, b, acc[1][j]);
      }
    }
    #pragma unroll
    for (int m = 0; m < 2; ++m)
      #pragma unroll
      for (int j = 0; j < 2; ++j) {
        const int col = (wave * 2 + j) * 16 + nlo;
        #pragma unroll
        for (int r = 0; r < 8; ++r) {
          const int M = r + hi8 + m * 16;
          float v = acc[m][j][r];
          smH1[(size_t)M * CH + col] = (bf16)(v > 0.0f ? v : 0.0f);
        }
      }
  }
  __syncthreads();

  // ---- layer 2: [32,128] @ [128,128] ----
  {
    v8f acc[2][2];
    #pragma unroll
    for (int j = 0; j < 2; ++j) {
      const float bv = b2[(wave * 2 + j) * 16 + nlo];
      #pragma unroll
      for (int m = 0; m < 2; ++m)
        #pragma unroll
        for (int r = 0; r < 8; ++r) acc[m][j][r] = bv;
    }
    #pragma unroll
    for (int kt = 0; kt < CH / 32; ++kt) {
      const int koff = kt * 32 + hi8;
      const v16bf a0 = load_afrag(rA2[0], koff);
      const v16bf a1 = load_afrag(rA2[1], koff);
      #pragma unroll
      for (int j = 0; j < 2; ++j) {
        const v16bf b = load_bfrag(w2f, kt * 8 + wave * 2 + j, lane);
        acc[0][j] = wmma_bf16(a0, b, acc[0][j]);
        acc[1][j] = wmma_bf16(a1, b, acc[1][j]);
      }
    }
    #pragma unroll
    for (int m = 0; m < 2; ++m)
      #pragma unroll
      for (int j = 0; j < 2; ++j) {
        const int col = (wave * 2 + j) * 16 + nlo;
        #pragma unroll
        for (int r = 0; r < 8; ++r) {
          const int M = r + hi8 + m * 16;
          float v = acc[m][j][r];
          smH2[(size_t)M * CH + col] = (bf16)(v > 0.0f ? v : 0.0f);
        }
      }
  }
  __syncthreads();

  // ---- layer 3: [32,128] @ [128,256]; scale by ew; atomic scatter ----
  {
    v16bf afr[2][CH / 32];
    #pragma unroll
    for (int m = 0; m < 2; ++m)
      #pragma unroll
      for (int kt = 0; kt < CH / 32; ++kt)
        afr[m][kt] = load_afrag(rA3[m], kt * 32 + hi8);
    #pragma unroll
    for (int j = 0; j < 4; ++j) {
      const int nt = wave * 4 + j;
      v8f acc[2];
      const float bv = b3[nt * 16 + nlo];
      #pragma unroll
      for (int m = 0; m < 2; ++m)
        #pragma unroll
        for (int r = 0; r < 8; ++r) acc[m][r] = bv;
      #pragma unroll
      for (int kt = 0; kt < CH / 32; ++kt) {
        const v16bf b = load_bfrag(w3f, kt * 16 + nt, lane);
        acc[0] = wmma_bf16(afr[0][kt], b, acc[0]);
        acc[1] = wmma_bf16(afr[1][kt], b, acc[1]);
      }
      #pragma unroll
      for (int m = 0; m < 2; ++m)
        #pragma unroll
        for (int r = 0; r < 8; ++r) {
          const int M = r + hi8 + m * 16;
          const float val = acc[m][r] * smEw[M];
          atomicAdd(agg + (size_t)smDst[M] * HDIM + nt * 16 + nlo, val);
        }
    }
  }
}

// ---------------------------------------------------------------------------
// Node update: joint = LN(s + agg/deg)*mask; latent_out; node_out = LN(...).
// 1 block (256 threads) per node.
// ---------------------------------------------------------------------------
__device__ __forceinline__ float block_sum(float v, float* red) {
  const int t = threadIdx.x;
  red[t] = v;
  __syncthreads();
  for (int sft = 128; sft > 0; sft >>= 1) {
    if (t < sft) red[t] += red[t + sft];
    __syncthreads();
  }
  const float r = red[0];
  __syncthreads();
  return r;
}

__launch_bounds__(256)
__global__ void node_update_kernel(const float* __restrict__ s,
                                   const float* __restrict__ agg,
                                   const float* __restrict__ deg,
                                   const float* __restrict__ mask,
                                   const float* __restrict__ latf,
                                   const float* __restrict__ nodef,
                                   const float* __restrict__ mlng, const float* __restrict__ mlnb,
                                   const float* __restrict__ latw, const float* __restrict__ latb,
                                   const float* __restrict__ nodew, const float* __restrict__ nodeb,
                                   const float* __restrict__ nlng, const float* __restrict__ nlnb,
                                   float* __restrict__ joint,
                                   float* __restrict__ lat_out,
                                   float* __restrict__ node_out) {
  __shared__ float red[256];
  __shared__ float sj[HDIM];
  __shared__ float sn[CS];
  const int n = blockIdx.x, t = threadIdx.x;

  const float d = fmaxf(deg[n], 1.0f);
  const float x = s[(size_t)n * HDIM + t] + agg[(size_t)n * HDIM + t] / d;
  const float m = block_sum(x, red) * (1.0f / HDIM);
  const float dx = x - m;
  const float var = block_sum(dx * dx, red) * (1.0f / HDIM);
  float j = dx * rsqrtf(var + 1e-5f) * mlng[t] + mlnb[t];
  j *= mask[n];
  sj[t] = j;
  joint[(size_t)n * HDIM + t] = j;
  __syncthreads();

  if (t < CLAT) {
    float a = latb[t];
    for (int k = 0; k < HDIM; ++k) a += sj[k] * latw[k * CLAT + t];
    lat_out[(size_t)n * CLAT + t] = latf[(size_t)n * CLAT + t] + a;
  } else if (t < CLAT + CS) {
    const int c = t - CLAT;
    float a = nodeb[c];
    for (int k = 0; k < HDIM; ++k) a += sj[k] * nodew[k * CS + c];
    sn[c] = nodef[(size_t)n * CS + c] + a;
  }
  __syncthreads();

  const float xv = (t < CS) ? sn[t] : 0.0f;
  const float m2 = block_sum(xv, red) * (1.0f / CS);
  const float dx2 = (t < CS) ? (sn[t] - m2) : 0.0f;
  const float v2 = block_sum(dx2 * dx2, red) * (1.0f / CS);
  if (t < CS)
    node_out[(size_t)n * CS + t] = dx2 * rsqrtf(v2 + 1e-5f) * nlng[t] + nlnb[t];
}

// ---------------------------------------------------------------------------
// Edge update: relu([joint[dst]|joint[src]|z] @ W1) @ W2; edge_out = LN(z + .)
// 1 block = 32 edges, 128 threads (4 waves), WMMA for both layers.
// ---------------------------------------------------------------------------
__launch_bounds__(128)
__global__ void edge_update_kernel(const float* __restrict__ joint,
                                   const float* __restrict__ ef,
                                   const int*   __restrict__ ei,
                                   const bf16*  __restrict__ w1f, const float* __restrict__ b1,
                                   const bf16*  __restrict__ w2f, const float* __restrict__ b2,
                                   const float* __restrict__ elng, const float* __restrict__ elnb,
                                   float* __restrict__ edge_out) {
  __shared__ alignas(16) bf16 smA[MTILE * EIN_DIM];
  __shared__ alignas(16) bf16 smH[MTILE * CH];
  __shared__ float smE[MTILE * CZ];

  const int tid  = threadIdx.x;
  const int lane = tid & 31;
  const int wave = tid >> 5;
  const int e0   = blockIdx.x * MTILE;

  // ---- stage e_in = [joint[dst] | joint[src] | z] (4 thr/edge) ----
  {
    const int le = tid >> 2, sub = tid & 3;
    const int e = e0 + le;
    const int src = ei[e], dst = ei[NEDGES + e];
    bf16* row = smA + (size_t)le * EIN_DIM;
    const float* jd = joint + (size_t)dst * HDIM;
    const float* js = joint + (size_t)src * HDIM;
    for (int i = sub * 64; i < sub * 64 + 64; ++i) {
      row[i]        = (bf16)jd[i];
      row[HDIM + i] = (bf16)js[i];
    }
    for (int i = sub * 16; i < sub * 16 + 16; ++i)
      row[2 * HDIM + i] = (bf16)ef[(size_t)e * CZ + i];
  }
  __syncthreads();

  const int nlo = lane & 15;
  const int hi8 = (lane >> 4) * 8;
  const bf16* rA1[2] = { smA + (size_t)nlo * EIN_DIM,
                         smA + (size_t)(nlo + 16) * EIN_DIM };
  const bf16* rA2[2] = { smH + (size_t)nlo * CH,
                         smH + (size_t)(nlo + 16) * CH };

  // ---- layer 1: [32,576] @ [576,128] ----
  {
    v8f acc[2][2];
    #pragma unroll
    for (int j = 0; j < 2; ++j) {
      const float bv = b1[(wave * 2 + j) * 16 + nlo];
      #pragma unroll
      for (int m = 0; m < 2; ++m)
        #pragma unroll
        for (int r = 0; r < 8; ++r) acc[m][j][r] = bv;
    }
    for (int kt = 0; kt < EIN_DIM / 32; ++kt) {
      const int koff = kt * 32 + hi8;
      const v16bf a0 = load_afrag(rA1[0], koff);
      const v16bf a1 = load_afrag(rA1[1], koff);
      #pragma unroll
      for (int j = 0; j < 2; ++j) {
        const v16bf b = load_bfrag(w1f, kt * 8 + wave * 2 + j, lane);
        acc[0][j] = wmma_bf16(a0, b, acc[0][j]);
        acc[1][j] = wmma_bf16(a1, b, acc[1][j]);
      }
    }
    #pragma unroll
    for (int m = 0; m < 2; ++m)
      #pragma unroll
      for (int j = 0; j < 2; ++j) {
        const int col = (wave * 2 + j) * 16 + nlo;
        #pragma unroll
        for (int r = 0; r < 8; ++r) {
          const int M = r + hi8 + m * 16;
          float v = acc[m][j][r];
          smH[(size_t)M * CH + col] = (bf16)(v > 0.0f ? v : 0.0f);
        }
      }
  }
  __syncthreads();

  // ---- layer 2: [32,128] @ [128,64], one n-tile per wave ----
  {
    v8f acc[2];
    const float bv = b2[wave * 16 + nlo];
    #pragma unroll
    for (int m = 0; m < 2; ++m)
      #pragma unroll
      for (int r = 0; r < 8; ++r) acc[m][r] = bv;
    #pragma unroll
    for (int kt = 0; kt < CH / 32; ++kt) {
      const int koff = kt * 32 + hi8;
      const v16bf a0 = load_afrag(rA2[0], koff);
      const v16bf a1 = load_afrag(rA2[1], koff);
      const v16bf b  = load_bfrag(w2f, kt * 4 + wave, lane);
      acc[0] = wmma_bf16(a0, b, acc[0]);
      acc[1] = wmma_bf16(a1, b, acc[1]);
    }
    #pragma unroll
    for (int m = 0; m < 2; ++m)
      #pragma unroll
      for (int r = 0; r < 8; ++r) {
        const int M = r + hi8 + m * 16;
        smE[(size_t)M * CZ + wave * 16 + nlo] = acc[m][r];
      }
  }
  __syncthreads();

  // ---- residual + LayerNorm over 64, one edge per thread (tid<32) ----
  if (tid < MTILE) {
    const int e = e0 + tid;
    float x[CZ];
    float sum = 0.0f;
    for (int c = 0; c < CZ; ++c) {
      x[c] = ef[(size_t)e * CZ + c] + smE[(size_t)tid * CZ + c];
      sum += x[c];
    }
    const float m = sum * (1.0f / CZ);
    float vs = 0.0f;
    for (int c = 0; c < CZ; ++c) { const float dd = x[c] - m; vs += dd * dd; }
    const float inv = rsqrtf(vs * (1.0f / CZ) + 1e-5f);
    for (int c = 0; c < CZ; ++c)
      edge_out[(size_t)e * CZ + c] = (x[c] - m) * inv * elng[c] + elnb[c];
  }
}

// ---------------------------------------------------------------------------
// Host launcher
// ---------------------------------------------------------------------------
extern "C" void kernel_launch(void* const* d_in, const int* in_sizes, int n_in,
                              void* d_out, int out_size, void* d_ws, size_t ws_size,
                              hipStream_t stream) {
  (void)in_sizes; (void)n_in; (void)out_size; (void)ws_size;
  const float* latf   = (const float*)d_in[0];
  const float* nodef  = (const float*)d_in[1];
  const float* ef     = (const float*)d_in[2];
  const int*   ei     = (const int*)  d_in[3];
  const float* rots   = (const float*)d_in[4];
  const float* trans  = (const float*)d_in[5];
  const float* mask   = (const float*)d_in[6];
  const float* pw     = (const float*)d_in[7];
  const float* pb     = (const float*)d_in[8];
  const float* mw1    = (const float*)d_in[9];
  const float* mb1    = (const float*)d_in[10];
  const float* mw2    = (const float*)d_in[11];
  const float* mb2    = (const float*)d_in[12];
  const float* mw3    = (const float*)d_in[13];
  const float* mb3    = (const float*)d_in[14];
  const float* mlng   = (const float*)d_in[15];
  const float* mlnb   = (const float*)d_in[16];
  const float* ew1    = (const float*)d_in[17];
  const float* eb1    = (const float*)d_in[18];
  const float* ew2    = (const float*)d_in[19];
  const float* eb2    = (const float*)d_in[20];
  const float* elng   = (const float*)d_in[21];
  const float* elnb   = (const float*)d_in[22];
  const float* latw   = (const float*)d_in[23];
  const float* latb   = (const float*)d_in[24];
  const float* nodew  = (const float*)d_in[25];
  const float* nodeb  = (const float*)d_in[26];
  const float* nlng   = (const float*)d_in[27];
  const float* nlnb   = (const float*)d_in[28];

  float* lat_out  = (float*)d_out;                                // [N,64]
  float* node_out = (float*)d_out + (size_t)NNODES * CLAT;        // [N,128]
  float* edge_out = (float*)d_out + (size_t)NNODES * (CLAT + CS); // [E,64]

  // workspace layout
  char* ws = (char*)d_ws;
  size_t off = 0;
  auto take = [&](size_t bytes) -> char* {
    char* p = ws + off;
    off = (off + bytes + 255) & ~(size_t)255;
    return p;
  };
  float* s     = (float*)take((size_t)NNODES * HDIM * 4);
  float* pts   = (float*)take((size_t)NNODES * PNT * 3 * 4);
  float* glob  = (float*)take((size_t)NNODES * PNT * 3 * 4);
  float* agg   = (float*)take((size_t)NNODES * HDIM * 4);
  float* deg   = (float*)take((size_t)NNODES * 4);
  float* joint = (float*)take((size_t)NNODES * HDIM * 4);
  bf16* mw1f = (bf16*)take((size_t)(MIN_DIM / 32) * 8  * 512 * 2);
  bf16* mw2f = (bf16*)take((size_t)(CH / 32)      * 8  * 512 * 2);
  bf16* mw3f = (bf16*)take((size_t)(CH / 32)      * 16 * 512 * 2);
  bf16* ew1f = (bf16*)take((size_t)(EIN_DIM / 32) * 8  * 512 * 2);
  bf16* ew2f = (bf16*)take((size_t)(CH / 32)      * 4  * 512 * 2);

  // 1) zero accumulators (agg then deg, contiguous in ws)
  {
    const int tot = NNODES * HDIM + NNODES;
    zero_kernel<<<(tot + 255) / 256, 256, 0, stream>>>(agg, tot);
  }
  // 2) repack weights to bf16 B-fragments
  wprep_kernel<<<dim3(8,  MIN_DIM / 32), 32, 0, stream>>>(mw1, mw1f, CH);
  wprep_kernel<<<dim3(8,  CH / 32),      32, 0, stream>>>(mw2, mw2f, CH);
  wprep_kernel<<<dim3(16, CH / 32),      32, 0, stream>>>(mw3, mw3f, HDIM);
  wprep_kernel<<<dim3(8,  EIN_DIM / 32), 32, 0, stream>>>(ew1, ew1f, CH);
  wprep_kernel<<<dim3(4,  CH / 32),      32, 0, stream>>>(ew2, ew2f, CZ);
  // 3) per-node prep
  node_prep_kernel<<<NNODES, 32, 0, stream>>>(latf, nodef, rots, trans, pw, pb,
                                              s, pts, glob);
  // 4) message MLP + scatter (WMMA)
  msg_kernel<<<NEDGES / MTILE, 128, 0, stream>>>(s, pts, glob, ef, ei, rots,
                                                 trans, mask, mw1f, mb1, mw2f,
                                                 mb2, mw3f, mb3, agg, deg);
  // 5) node update (joint, latent_out, node_out)
  node_update_kernel<<<NNODES, 256, 0, stream>>>(s, agg, deg, mask, latf, nodef,
                                                 mlng, mlnb, latw, latb, nodew,
                                                 nodeb, nlng, nlnb, joint,
                                                 lat_out, node_out);
  // 6) edge update (WMMA) -> edge_out
  edge_update_kernel<<<NEDGES / MTILE, 128, 0, stream>>>(joint, ef, ei, ew1f,
                                                         eb1, ew2f, eb2, elng,
                                                         elnb, edge_out);
}
